// CAGetBoard_1623497638222
// MI455X (gfx1250) — matrix operand order repaired and checked by reading, not compiled, not run on hardware
//
#include <hip/hip_runtime.h>
#include <cstddef>

typedef __attribute__((ext_vector_type(16))) _Float16 v16h;
typedef __attribute__((ext_vector_type(8)))  _Float16 v8h;
typedef __attribute__((ext_vector_type(8)))  float    v8f;

#define IMG_H 256
#define IMG_W 256
#define NCH   16
#define TILE_W 16
#define TILE_H 16   // 16x16 pixel tile per block -> 256 px -> 8 waves x 2 M-tiles

// ---------------------------------------------------------------------------
// Pass 0: one-time (per launch) weight conversion to f16 in scratch.
// w1h: [128][64] f16 (K 48 zero-padded to 64), w2h: [16][128] f16.
// ---------------------------------------------------------------------------
__global__ __launch_bounds__(256) void nca_pass0(
    const float* __restrict__ w1, const float* __restrict__ w2,
    _Float16* __restrict__ w1h, _Float16* __restrict__ w2h)
{
    int i = blockIdx.x * 256 + threadIdx.x;
    if (i < 128 * 64) {
        int o = i >> 6, k = i & 63;
        w1h[i] = (k < 48) ? (_Float16)w1[o * 48 + k] : (_Float16)0.f;
    }
    int j = i - 128 * 64;
    if (j >= 0 && j < 16 * 128)
        w2h[j] = (_Float16)w2[j];
}

// ---------------------------------------------------------------------------
// Pass 1: perception + MLP (WMMA) + stochastic update. Writes boards_pre to
// out (all 16 ch) and channel 3 additionally to ch3buf (for pass-2 maxpool).
// ---------------------------------------------------------------------------
__global__ __launch_bounds__(256) void nca_pass1(
    const float* __restrict__ x,
    const _Float16* __restrict__ w1h, const float* __restrict__ b1,
    const _Float16* __restrict__ w2h, const float* __restrict__ b2,
    const float* __restrict__ rmask,
    float* __restrict__ out, float* __restrict__ ch3buf)
{
    __shared__ float xs[NCH][TILE_H + 2][TILE_W + 2];             // x tile + halo
    __shared__ __align__(16) _Float16 perc[256][64];              // [pixel][K] (48 pad 64)
    __shared__ __align__(16) _Float16 hbuf[8][16][128];           // per-wave hidden
    __shared__ __align__(16) _Float16 w1s[128][64];               // w1 as [out][K] f16
    __shared__ __align__(16) _Float16 w2s[16][128];               // w2 as [out][K] f16
    __shared__ float b1s[128];
    __shared__ float b2s[16];
    __shared__ float rms[256];                                    // stochastic mask

    const int tid = threadIdx.x;
    const int tx = blockIdx.x, ty = blockIdx.y, b = blockIdx.z;
    const int x0 = tx * TILE_W, y0 = ty * TILE_H;

    // ---- load x tile with 1-px halo (zero padded at image borders) ----
    for (int i = tid; i < NCH * 18 * 18; i += 256) {
        int ch = i / 324, p = i % 324, hr = p / 18, hc = p % 18;
        int yy = y0 + hr - 1, xx = x0 + hc - 1;
        float v = 0.f;
        if (yy >= 0 && yy < IMG_H && xx >= 0 && xx < IMG_W)
            v = x[(((size_t)b * NCH + ch) * IMG_H + yy) * IMG_W + xx];
        xs[ch][hr][hc] = v;
    }
    // ---- stage pre-converted f16 weights via 16B vector copies ----
    {
        v8h* dst = (v8h*)&w1s[0][0];
        const v8h* src = (const v8h*)w1h;
        for (int i = tid; i < (128 * 64) / 8; i += 256) dst[i] = src[i];
        v8h* dst2 = (v8h*)&w2s[0][0];
        const v8h* src2 = (const v8h*)w2h;
        if (tid < (16 * 128) / 8) dst2[tid] = src2[tid];
    }
    if (tid < 128) b1s[tid] = b1[tid];
    if (tid < 16)  b2s[tid] = b2[tid];
    {
        int r = tid >> 4, c = tid & 15;
        rms[tid] = (rmask[((size_t)b * IMG_H + y0 + r) * IMG_W + x0 + c] < 0.5f) ? 1.f : 0.f;
    }
    __syncthreads();

    // ---- perception: identity + sobel-x + sobel-y, packed f16 [pixel][64] ----
    {
        int pix = tid;
        int r = pix >> 4, c = pix & 15;
        for (int ch = 0; ch < NCH; ++ch) {
            float m00 = xs[ch][r][c],     m01 = xs[ch][r][c + 1],     m02 = xs[ch][r][c + 2];
            float m10 = xs[ch][r + 1][c],                             m12 = xs[ch][r + 1][c + 2];
            float m20 = xs[ch][r + 2][c], m21 = xs[ch][r + 2][c + 1], m22 = xs[ch][r + 2][c + 2];
            float m11 = xs[ch][r + 1][c + 1];
            float sgx = -m00 + m02 - 2.f * m10 + 2.f * m12 - m20 + m22;
            float sgy = -m00 - 2.f * m01 - m02 + m20 + 2.f * m21 + m22;
            perc[pix][ch]      = (_Float16)m11;
            perc[pix][16 + ch] = (_Float16)sgx;
            perc[pix][32 + ch] = (_Float16)sgy;
        }
        for (int k = 48; k < 64; ++k) perc[pix][k] = (_Float16)0.f;
    }
    __syncthreads();

    // ---- WMMA MLP. Each wave: two 16-pixel M-tiles, GEMM1 -> GEMM2 fused. ----
    const int lane  = tid & 31;
    const int wv    = tid >> 5;       // wave 0..7
    const int nl    = lane & 15;      // N (or M row) lane index
    const int hl    = lane >> 4;      // lane half-group

    for (int half = 0; half < 2; ++half) {
        const int mb   = wv * 32 + half * 16;   // first pixel of this M-tile
        const int pixA = mb + nl;               // A-matrix row (pixel) for this lane

        // A operand: perception, K chunks [0..31] and [32..63]
        v16h a0, a1;
        {
            union { v16h v; v8h h[2]; } u;
            u.h[0] = *(const v8h*)&perc[pixA][hl * 8];
            u.h[1] = *(const v8h*)&perc[pixA][16 + hl * 8];
            a0 = u.v;
            u.h[0] = *(const v8h*)&perc[pixA][32 + hl * 8];
            u.h[1] = *(const v8h*)&perc[pixA][48 + hl * 8];
            a1 = u.v;
        }

        // GEMM1: [16 x 48] x [48 x 128] -> hidden, 8 N-tiles x 2 WMMAs
        for (int t = 0; t < 8; ++t) {
            v16h bA, bB;
            {
                union { v16h v; v8h h[2]; } u;
                const _Float16* bp = &w1s[t * 16 + nl][hl * 8];
                u.h[0] = *(const v8h*)(bp);
                u.h[1] = *(const v8h*)(bp + 16);
                bA = u.v;
                bp = &w1s[t * 16 + nl][32 + hl * 8];
                u.h[0] = *(const v8h*)(bp);
                u.h[1] = *(const v8h*)(bp + 16);
                bB = u.v;
            }
            v8f acc;
            float bias = b1s[t * 16 + nl];
            for (int v = 0; v < 8; ++v) acc[v] = bias;
            acc = __builtin_amdgcn_wmma_f32_16x16x32_f16(false, a0, false, bA, (short)0, acc, false, false);
            acc = __builtin_amdgcn_wmma_f32_16x16x32_f16(false, a1, false, bB, (short)0, acc, false, false);
            // ReLU, park hidden in this wave's LDS slot (C-layout -> A-layout)
            for (int v = 0; v < 8; ++v) {
                float hv = acc[v] > 0.f ? acc[v] : 0.f;
                int m = v + 8 * hl;
                hbuf[wv][m][t * 16 + nl] = (_Float16)hv;
            }
        }
        // same-wave LDS RAW (cross-lane): wait for DS stores to land
        asm volatile("s_wait_dscnt 0" ::: "memory");

        // GEMM2: [16 x 128] x [128 x 16] -> dboard, 4 WMMAs
        v16h h0, h1, h2, h3;
        {
            union { v16h v; v8h h[2]; } u;
            const _Float16* hp = &hbuf[wv][nl][0];
            u.h[0] = *(const v8h*)(hp + hl * 8);       u.h[1] = *(const v8h*)(hp + 16 + hl * 8);  h0 = u.v;
            u.h[0] = *(const v8h*)(hp + 32 + hl * 8);  u.h[1] = *(const v8h*)(hp + 48 + hl * 8);  h1 = u.v;
            u.h[0] = *(const v8h*)(hp + 64 + hl * 8);  u.h[1] = *(const v8h*)(hp + 80 + hl * 8);  h2 = u.v;
            u.h[0] = *(const v8h*)(hp + 96 + hl * 8);  u.h[1] = *(const v8h*)(hp + 112 + hl * 8); h3 = u.v;
        }
        v16h g0, g1, g2, g3;
        {
            union { v16h v; v8h h[2]; } u;
            const _Float16* gp = &w2s[nl][0];
            u.h[0] = *(const v8h*)(gp + hl * 8);       u.h[1] = *(const v8h*)(gp + hl * 8 + 16);       g0 = u.v;
            u.h[0] = *(const v8h*)(gp + 32 + hl * 8);  u.h[1] = *(const v8h*)(gp + 32 + hl * 8 + 16);  g1 = u.v;
            u.h[0] = *(const v8h*)(gp + 64 + hl * 8);  u.h[1] = *(const v8h*)(gp + 64 + hl * 8 + 16);  g2 = u.v;
            u.h[0] = *(const v8h*)(gp + 96 + hl * 8);  u.h[1] = *(const v8h*)(gp + 96 + hl * 8 + 16);  g3 = u.v;
        }
        v8f acc2;
        {
            float bias2 = b2s[nl];
            for (int v = 0; v < 8; ++v) acc2[v] = bias2;
        }
        acc2 = __builtin_amdgcn_wmma_f32_16x16x32_f16(false, h0, false, g0, (short)0, acc2, false, false);
        acc2 = __builtin_amdgcn_wmma_f32_16x16x32_f16(false, h1, false, g1, (short)0, acc2, false, false);
        acc2 = __builtin_amdgcn_wmma_f32_16x16x32_f16(false, h2, false, g2, (short)0, acc2, false, false);
        acc2 = __builtin_amdgcn_wmma_f32_16x16x32_f16(false, h3, false, g3, (short)0, acc2, false, false);

        // ---- epilogue: tanh, stochastic mask, add state, store boards_pre ----
        for (int v = 0; v < 8; ++v) {
            int m   = v + 8 * hl;
            int pix = mb + m;
            int r = pix >> 4, c = pix & 15;
            float d = tanhf(acc2[v]) * rms[pix];
            float board = xs[nl][r + 1][c + 1] + d;
            int yy = y0 + r, xx = x0 + c;
            size_t oidx = (((size_t)b * NCH + nl) * IMG_H + yy) * IMG_W + xx;
            out[oidx] = board;
            if (nl == 3)
                ch3buf[((size_t)b * IMG_H + yy) * IMG_W + xx] = board;
        }
    }
}

// ---------------------------------------------------------------------------
// Pass 2: alive masking (pre from x ch3, post from ch3buf) + clip ch0-2.
// In-place RMW of out, each thread touches only its own pixel.
// ---------------------------------------------------------------------------
__global__ __launch_bounds__(256) void nca_pass2(
    const float* __restrict__ x, const float* __restrict__ ch3buf,
    float* __restrict__ out)
{
    size_t idx = (size_t)blockIdx.x * blockDim.x + threadIdx.x;
    const size_t NP = (size_t)32 * IMG_H * IMG_W;
    if (idx >= NP) return;
    int xx = (int)(idx % IMG_W);
    int yy = (int)((idx / IMG_W) % IMG_H);
    int b  = (int)(idx / ((size_t)IMG_H * IMG_W));

    const float* x3 = x      + (((size_t)b * NCH) + 3) * IMG_H * IMG_W;
    const float* p3 = ch3buf + (size_t)b * IMG_H * IMG_W;
    float pre = -1e30f, post = -1e30f;
    for (int dy = -1; dy <= 1; ++dy) {
        int ry = yy + dy;
        if (ry < 0 || ry >= IMG_H) continue;
        for (int dx = -1; dx <= 1; ++dx) {
            int rx = xx + dx;
            if (rx < 0 || rx >= IMG_W) continue;
            float a = x3[(size_t)ry * IMG_W + rx]; pre  = a > pre  ? a : pre;
            float q = p3[(size_t)ry * IMG_W + rx]; post = q > post ? q : post;
        }
    }
    float alive = (pre > 0.1f && post > 0.1f) ? 1.f : 0.f;

    float* ob = out + (size_t)b * NCH * IMG_H * IMG_W + (size_t)yy * IMG_W + xx;
    for (int ch = 0; ch < NCH; ++ch) {
        float v = ob[(size_t)ch * IMG_H * IMG_W] * alive;
        if (ch < 3) v = v < 0.f ? 0.f : (v > 1.f ? 1.f : v);
        ob[(size_t)ch * IMG_H * IMG_W] = v;
    }
}

// ---------------------------------------------------------------------------
extern "C" void kernel_launch(void* const* d_in, const int* in_sizes, int n_in,
                              void* d_out, int out_size, void* d_ws, size_t ws_size,
                              hipStream_t stream) {
    const float* x  = (const float*)d_in[0];
    const float* w1 = (const float*)d_in[1];
    const float* b1 = (const float*)d_in[2];
    const float* w2 = (const float*)d_in[3];
    const float* b2 = (const float*)d_in[4];
    const float* rm = (const float*)d_in[5];
    float* out = (float*)d_out;

    // scratch layout: [B*H*W f32 ch3 buffer][128*64 f16 w1][16*128 f16 w2]
    float*     ch3 = (float*)d_ws;
    _Float16*  w1h = (_Float16*)((char*)d_ws + (size_t)32 * IMG_H * IMG_W * sizeof(float));
    _Float16*  w2h = w1h + 128 * 64;

    nca_pass0<<<(128 * 64 + 16 * 128 + 255) / 256, 256, 0, stream>>>(w1, w2, w1h, w2h);

    dim3 grid(IMG_W / TILE_W, IMG_H / TILE_H, 32);
    nca_pass1<<<grid, 256, 0, stream>>>(x, w1h, b1, w2h, b2, rm, out, ch3);

    size_t np = (size_t)32 * IMG_H * IMG_W;
    nca_pass2<<<(np + 255) / 256, 256, 0, stream>>>(x, ch3, out);
}
